// MultiVAE_56006373539982
// MI455X (gfx1250) — matrix operand order repaired and checked by reading, not compiled, not run on hardware
//
#include <hip/hip_runtime.h>

typedef __attribute__((ext_vector_type(16))) __bf16 v16bf;
typedef __attribute__((ext_vector_type(8)))  float  v8f;

#define HALF_LOG_2PI 0.9189385332046727f

// ---- workspace layout (float elements), per-expert region ----
#define OFF_ACT1 0          // 128*1280
#define OFF_ACT2 163840     // 128*1280
#define OFF_C    327680     // 128*160  (xe cols 0..127, te cols 128..159)
#define OFF_ENC  348160     // 128*256  (mu_q, ls_q)
#define OFF_Z    380928     // 128*128
#define OFF_INP  397312     // 128*288  ([z, c])
#define OFF_MH1  434176     // 128*256
#define OFF_MH2  466944     // 128*256
#define OFF_MOUT 499712     // 128*256
#define OFF_DEC  532480     // 128*512
#define OFF_LOGDET 598016   // 128
#define OFF_LOGQ   598144   // 128
#define OFF_LOGP   598272   // 128
#define OFF_ELBO   598400   // 128
#define EXP_WS     598528   // floats per expert

#define LDSTR 40            // LDS row stride in bf16 halves (80B, 16B-aligned, conflict-spread)

__device__ __forceinline__ unsigned short f2bf(float f){
  unsigned int u = __float_as_uint(f);
  u += 0x7FFFu + ((u >> 16) & 1u);           // round-to-nearest-even
  return (unsigned short)(u >> 16);
}

__device__ __forceinline__ float silu_f(float x){ return x * (1.0f / (1.0f + __expf(-x))); }

// MADE mask computed from indices: modes
// 0=dense, 1=m1(order=asc), 2=m1(order=desc), 3=m2, 4=m3(asc), 5=m3(desc)
__device__ __forceinline__ bool wkeep(int mode, int n, int k){
  switch(mode){
    default: return true;
    case 1: return (k >= 128) || ((n % 127) >= k);
    case 2: return (k >= 128) || ((n % 127) >= (127 - k));
    case 3: return (n % 127) >= (k % 127);
    case 4: return (n & 127) > (k % 127);
    case 5: return (127 - (n & 127)) > (k % 127);
  }
}

union Frag { v16bf v; uint4 q[2]; };

// Out[b, n] = act( sum_k A[b,k] * W[n,k] (masked) + bias[n] ), B=128 rows fixed.
// grid.x tiles N by 128, grid.y = expert. 256 threads = 8 wave32.
__global__ __launch_bounds__(256) void gemm_bf16_kernel(
    const float* __restrict__ A, long long aStr, int lda,
    const float* __restrict__ W, long long wStr,
    const float* __restrict__ bias, long long bStr,
    float* __restrict__ Out, long long oStr, int ldo, int ocol,
    int K, int act, int mmode)
{
  const int e   = blockIdx.y;
  const int n0  = blockIdx.x * 128;
  const int tid = threadIdx.x;
  __shared__ unsigned short lA[128 * LDSTR];
  __shared__ unsigned short lW[128 * LDSTR];

  A    += (long long)e * aStr;
  W    += (long long)e * wStr;
  bias += (long long)e * bStr;
  Out  += (long long)e * oStr + ocol;

  const int row   = tid >> 1;      // 0..127 (tile row loaded by this thread)
  const int halfq = tid & 1;       // which 16-col half of the 32-wide K slab
  const int lane  = tid & 31;
  const int wave  = tid >> 5;
  const int mw    = wave & 3;      // M block of 32 rows
  const int nw    = wave >> 2;     // N block of 64 cols
  const int hi    = (lane >= 16);
  const int lcol  = lane & 15;

  v8f acc[2][4];
  const v8f zero8 = {0.f,0.f,0.f,0.f,0.f,0.f,0.f,0.f};
  #pragma unroll
  for (int mi = 0; mi < 2; ++mi)
    #pragma unroll
    for (int ni = 0; ni < 4; ++ni) acc[mi][ni] = zero8;

  for (int k0 = 0; k0 < K; k0 += 32) {
    { // A tile [128 x 32] f32 -> bf16 LDS
      const float4* src = (const float4*)(A + (long long)row * lda + k0 + halfq * 16);
      unsigned int* dst = (unsigned int*)&lA[row * LDSTR + halfq * 16];
      #pragma unroll
      for (int i = 0; i < 4; ++i) {
        float4 f = src[i];
        dst[2*i+0] = (unsigned int)f2bf(f.x) | ((unsigned int)f2bf(f.y) << 16);
        dst[2*i+1] = (unsigned int)f2bf(f.z) | ((unsigned int)f2bf(f.w) << 16);
      }
    }
    { // W tile [128 x 32] f32 -> (mask) -> bf16 LDS
      const int n = n0 + row;
      const float4* src = (const float4*)(W + (long long)n * K + k0 + halfq * 16);
      unsigned int* dst = (unsigned int*)&lW[row * LDSTR + halfq * 16];
      #pragma unroll
      for (int i = 0; i < 4; ++i) {
        float4 f = src[i];
        if (mmode) {
          const int kg = k0 + halfq * 16 + i * 4;
          if (!wkeep(mmode, n, kg+0)) f.x = 0.f;
          if (!wkeep(mmode, n, kg+1)) f.y = 0.f;
          if (!wkeep(mmode, n, kg+2)) f.z = 0.f;
          if (!wkeep(mmode, n, kg+3)) f.w = 0.f;
        }
        dst[2*i+0] = (unsigned int)f2bf(f.x) | ((unsigned int)f2bf(f.y) << 16);
        dst[2*i+1] = (unsigned int)f2bf(f.z) | ((unsigned int)f2bf(f.w) << 16);
      }
    }
    __syncthreads();

    Frag a[2], b[4];
    #pragma unroll
    for (int mi = 0; mi < 2; ++mi) {  // A frag: lane row = M, halves K kb..kb+7 and kb+16..kb+23
      const int r = mw * 32 + mi * 16 + lcol;
      const uint4* p = (const uint4*)&lA[r * LDSTR + (hi ? 8 : 0)];
      a[mi].q[0] = p[0];
      a[mi].q[1] = p[2];
    }
    #pragma unroll
    for (int ni = 0; ni < 4; ++ni) {  // B frag: lane col = N, halves K kb..kb+7 and kb+16..kb+23
      const int r = nw * 64 + ni * 16 + lcol;
      const uint4* p = (const uint4*)&lW[r * LDSTR + (hi ? 8 : 0)];
      b[ni].q[0] = p[0];
      b[ni].q[1] = p[2];
    }
    #pragma unroll
    for (int mi = 0; mi < 2; ++mi)
      #pragma unroll
      for (int ni = 0; ni < 4; ++ni)
        acc[mi][ni] = __builtin_amdgcn_wmma_f32_16x16x32_bf16(
            false, a[mi].v, false, b[ni].v, (short)0, acc[mi][ni], false, false);
    __syncthreads();
  }

  // epilogue: bias + activation + store
  #pragma unroll
  for (int ni = 0; ni < 4; ++ni) {
    const int col = n0 + nw * 64 + ni * 16 + lcol;
    const float bs = bias[col];
    #pragma unroll
    for (int mi = 0; mi < 2; ++mi) {
      #pragma unroll
      for (int r = 0; r < 8; ++r) {
        const int rw = mw * 32 + mi * 16 + r + (hi ? 8 : 0);
        float v = acc[mi][ni][r] + bs;
        if (act == 1)      v = fmaxf(v, 0.f);
        else if (act == 2) v = silu_f(v);
        Out[(long long)rw * ldo + col] = v;
      }
    }
  }
}

// t-embedding MLP: [B,1] -> 256 -> 256 -> 32 (silu,silu,silu). grid(16 experts, 8 sample-groups)
__global__ __launch_bounds__(256) void tmlp_kernel(
    const float* __restrict__ t,
    const float* __restrict__ tw1, const float* __restrict__ tb1,
    const float* __restrict__ tw2, const float* __restrict__ tb2,
    const float* __restrict__ tw3, const float* __restrict__ tb3,
    float* __restrict__ ws)
{
  const int e = blockIdx.x, g = blockIdx.y;
  const int j = threadIdx.x;
  __shared__ float h1[256][16];
  __shared__ float h2[256][16];
  float* c = ws + (long long)e * EXP_WS + OFF_C;

  {
    const float w = tw1[e * 256 + j], b = tb1[e * 256 + j];
    #pragma unroll
    for (int s = 0; s < 16; ++s) h1[j][s] = silu_f(w * t[g * 16 + s] + b);
  }
  __syncthreads();
  {
    const float* wr = tw2 + ((long long)e * 256 + j) * 256;
    float acc[16];
    #pragma unroll
    for (int s = 0; s < 16; ++s) acc[s] = 0.f;
    for (int k = 0; k < 256; ++k) {
      const float w = wr[k];
      #pragma unroll
      for (int s = 0; s < 16; ++s) acc[s] += w * h1[k][s];
    }
    const float b = tb2[e * 256 + j];
    #pragma unroll
    for (int s = 0; s < 16; ++s) h2[j][s] = silu_f(acc[s] + b);
  }
  __syncthreads();
  if (j < 32) {
    const float* wr = tw3 + ((long long)e * 32 + j) * 256;
    float acc[16];
    #pragma unroll
    for (int s = 0; s < 16; ++s) acc[s] = 0.f;
    for (int k = 0; k < 256; ++k) {
      const float w = wr[k];
      #pragma unroll
      for (int s = 0; s < 16; ++s) acc[s] += w * h2[k][s];
    }
    const float b = tb3[e * 32 + j];
    #pragma unroll
    for (int s = 0; s < 16; ++s)
      c[(long long)(g * 16 + s) * 160 + 128 + j] = silu_f(acc[s] + b);
  }
}

// z = mu + exp(ls)*eps ; log_q ; build inp=[z,c] ; logdet=0
__global__ __launch_bounds__(128) void sample_kernel(
    const float* __restrict__ eps, float* __restrict__ ws)
{
  const int e = blockIdx.x, b = threadIdx.x;
  float* base = ws + (long long)e * EXP_WS;
  const float* enc = base + OFF_ENC + b * 256;
  const float* c   = base + OFF_C   + b * 160;
  float* z   = base + OFF_Z   + b * 128;
  float* inp = base + OFF_INP + b * 288;
  float lq = 0.f;
  for (int h = 0; h < 128; ++h) {
    const float mu = enc[h], ls = enc[128 + h];
    const float zz = mu + __expf(ls) * eps[b * 128 + h];
    z[h] = zz; inp[h] = zz;
    const float u = (zz - mu) * __expf(-ls);
    lq += -0.5f * u * u - ls - HALF_LOG_2PI;
  }
  base[OFF_LOGQ + b] = lq;
  base[OFF_LOGDET + b] = 0.f;
  for (int j = 0; j < 160; ++j) inp[128 + j] = c[j];
}

// MAF per-transform post: z=(z-mu)*exp(-ls); logdet-=sum(ls); optional final log_prior
__global__ __launch_bounds__(128) void maf_post_kernel(float* __restrict__ ws, int finalk)
{
  const int e = blockIdx.x, b = threadIdx.x;
  float* base = ws + (long long)e * EXP_WS;
  const float* mo = base + OFF_MOUT + b * 256;
  float* z   = base + OFF_Z   + b * 128;
  float* inp = base + OFF_INP + b * 288;
  float ld = 0.f, s2 = 0.f;
  for (int h = 0; h < 128; ++h) {
    const float mu = mo[h], ls = mo[128 + h];
    const float zz = (z[h] - mu) * __expf(-ls);
    z[h] = zz; inp[h] = zz;
    ld -= ls; s2 += zz * zz;
  }
  const float lt = base[OFF_LOGDET + b] + ld;
  base[OFF_LOGDET + b] = lt;
  if (finalk) base[OFF_LOGP + b] = -0.5f * s2 - 128.0f * HALF_LOG_2PI + lt;
}

// per-expert elbo = gauss_lp(v_t; mu_d, ls_d) + log_prior - log_q
__global__ __launch_bounds__(128) void elbo_kernel(
    const float* __restrict__ v_t, float* __restrict__ ws)
{
  const int e = blockIdx.x, b = threadIdx.x;
  float* base = ws + (long long)e * EXP_WS;
  const float* d = base + OFF_DEC + b * 512;
  float acc = 0.f;
  for (int f = 0; f < 256; ++f) {
    const float mu = d[f], ls = d[256 + f];
    const float u = (v_t[b * 256 + f] - mu) * __expf(-ls);
    acc += -0.5f * u * u - ls - HALF_LOG_2PI;
  }
  base[OFF_ELBO + b] = acc + base[OFF_LOGP + b] - base[OFF_LOGQ + b];
}

// time-bin select (searchsorted left over 17 bins) + in-range gate
__global__ __launch_bounds__(128) void select_kernel(
    const float* __restrict__ t, const float* __restrict__ ws, float* __restrict__ out)
{
  const int b = threadIdx.x;
  float bins[17];
  #pragma unroll
  for (int i = 0; i < 7; ++i) bins[i]      = (float)(0.15 * (double)i / 7.0);
  #pragma unroll
  for (int i = 0; i < 6; ++i) bins[7 + i]  = (float)(0.15 + 0.35 * (double)i / 6.0);
  #pragma unroll
  for (int i = 0; i < 3; ++i) bins[13 + i] = (float)(0.5 + 0.5 * (double)i / 3.0);
  bins[16] = 1.0f;
  const float tv = t[b];
  int idx = 0;
  #pragma unroll
  for (int i = 0; i < 17; ++i) idx += (bins[i] < tv) ? 1 : 0;
  int bid = idx - 1; if (bid < 0) bid = 0; if (bid > 15) bid = 15;
  const bool inr = (tv > bins[0]) && (tv <= bins[16]);
  out[b] = inr ? ws[(long long)bid * EXP_WS + OFF_ELBO + b] : 0.f;
}

extern "C" void kernel_launch(void* const* d_in, const int* in_sizes, int n_in,
                              void* d_out, int out_size, void* d_ws, size_t ws_size,
                              hipStream_t stream)
{
  (void)in_sizes; (void)out_size; (void)ws_size;
  const float* x_t = (const float*)d_in[0];
  const float* v_t = (const float*)d_in[1];
  const float* t   = (const float*)d_in[2];
  const float* eps = (const float*)d_in[3];

  // sorted-key param order; sizes include the leading NEXP=16 dim
  static const long long PSZ[48] = {
    20480,20480,8192, 2621440,26214400,10485760,           // db1..3, dw1..3
    20480,20480,4096, 5242880,26214400,5242880,            // eb1..3, ew1..3
    4096,4096,4096,4096, 4096,4096,4096,4096, 4096,4096,4096,4096, // mb1_*,mb2_*,mb3_*
    1179648,1179648,1179648,1179648,                       // mw1_*
    1048576,1048576,1048576,1048576,                       // mw2_*
    1048576,1048576,1048576,1048576,                       // mw3_*
    4096,4096,512, 4096,1048576,131072,                    // tb1..3, tw1..3
    20480,20480,2048, 5242880,26214400,2621440 };          // xb1..3, xw1..3
  const float* P[48];
  if (n_in >= 52) {
    for (int i = 0; i < 48; ++i) P[i] = (const float*)d_in[4 + i];
  } else {
    const float* base = (const float*)d_in[4];
    long long off = 0;
    for (int i = 0; i < 48; ++i) { P[i] = base + off; off += PSZ[i]; }
  }
  const float *db1=P[0], *db2=P[1], *db3=P[2], *dw1=P[3], *dw2=P[4], *dw3=P[5];
  const float *eb1=P[6], *eb2=P[7], *eb3=P[8], *ew1=P[9], *ew2=P[10], *ew3=P[11];
  const float *mb1[4]={P[12],P[13],P[14],P[15]};
  const float *mb2[4]={P[16],P[17],P[18],P[19]};
  const float *mb3[4]={P[20],P[21],P[22],P[23]};
  const float *mw1[4]={P[24],P[25],P[26],P[27]};
  const float *mw2[4]={P[28],P[29],P[30],P[31]};
  const float *mw3[4]={P[32],P[33],P[34],P[35]};
  const float *tb1=P[36], *tb2=P[37], *tb3=P[38], *tw1=P[39], *tw2=P[40], *tw3=P[41];
  const float *xb1=P[42], *xb2=P[43], *xb3=P[44], *xw1=P[45], *xw2=P[46], *xw3=P[47];

  float* ws = (float*)d_ws;
  float* out = (float*)d_out;
  const dim3 blk(256);

  // x-embedding: 256 -> 1280 -> 1280 -> 128 (silu x3), result into c[:,0:128]
  gemm_bf16_kernel<<<dim3(10,16), blk, 0, stream>>>(x_t, 0, 256, xw1, 327680, xb1, 1280,
      ws + OFF_ACT1, EXP_WS, 1280, 0, 256, 2, 0);
  gemm_bf16_kernel<<<dim3(10,16), blk, 0, stream>>>(ws + OFF_ACT1, EXP_WS, 1280, xw2, 1638400, xb2, 1280,
      ws + OFF_ACT2, EXP_WS, 1280, 0, 1280, 2, 0);
  gemm_bf16_kernel<<<dim3(1,16), blk, 0, stream>>>(ws + OFF_ACT2, EXP_WS, 1280, xw3, 163840, xb3, 128,
      ws + OFF_C, EXP_WS, 160, 0, 1280, 2, 0);

  // t-embedding into c[:,128:160]
  tmlp_kernel<<<dim3(16,8), blk, 0, stream>>>(t, tw1, tb1, tw2, tb2, tw3, tb3, ws);

  // encoder: v_t 256 -> 1280 -> 1280 -> 256 (relu,relu,none)
  gemm_bf16_kernel<<<dim3(10,16), blk, 0, stream>>>(v_t, 0, 256, ew1, 327680, eb1, 1280,
      ws + OFF_ACT1, EXP_WS, 1280, 0, 256, 1, 0);
  gemm_bf16_kernel<<<dim3(10,16), blk, 0, stream>>>(ws + OFF_ACT1, EXP_WS, 1280, ew2, 1638400, eb2, 1280,
      ws + OFF_ACT2, EXP_WS, 1280, 0, 1280, 1, 0);
  gemm_bf16_kernel<<<dim3(2,16), blk, 0, stream>>>(ws + OFF_ACT2, EXP_WS, 1280, ew3, 327680, eb3, 256,
      ws + OFF_ENC, EXP_WS, 256, 0, 1280, 0, 0);

  // rsample + log_q + build inp=[z,c]
  sample_kernel<<<dim3(16), dim3(128), 0, stream>>>(eps, ws);

  // 4 MAF transforms (MADE masks applied analytically inside the GEMM)
  for (int k = 0; k < 4; ++k) {
    const int m1 = (k % 2 == 0) ? 1 : 2;
    const int m3 = (k % 2 == 0) ? 4 : 5;
    gemm_bf16_kernel<<<dim3(2,16), blk, 0, stream>>>(ws + OFF_INP, EXP_WS, 288, mw1[k], 73728, mb1[k], 256,
        ws + OFF_MH1, EXP_WS, 256, 0, 288, 1, m1);
    gemm_bf16_kernel<<<dim3(2,16), blk, 0, stream>>>(ws + OFF_MH1, EXP_WS, 256, mw2[k], 65536, mb2[k], 256,
        ws + OFF_MH2, EXP_WS, 256, 0, 256, 1, 3);
    gemm_bf16_kernel<<<dim3(2,16), blk, 0, stream>>>(ws + OFF_MH2, EXP_WS, 256, mw3[k], 65536, mb3[k], 256,
        ws + OFF_MOUT, EXP_WS, 256, 0, 256, 0, m3);
    maf_post_kernel<<<dim3(16), dim3(128), 0, stream>>>(ws, (k == 3) ? 1 : 0);
  }

  // decoder: z 128 -> 1280 -> 1280 -> 512 (relu,relu,none)
  gemm_bf16_kernel<<<dim3(10,16), blk, 0, stream>>>(ws + OFF_Z, EXP_WS, 128, dw1, 163840, db1, 1280,
      ws + OFF_ACT1, EXP_WS, 1280, 0, 128, 1, 0);
  gemm_bf16_kernel<<<dim3(10,16), blk, 0, stream>>>(ws + OFF_ACT1, EXP_WS, 1280, dw2, 1638400, db2, 1280,
      ws + OFF_ACT2, EXP_WS, 1280, 0, 1280, 1, 0);
  gemm_bf16_kernel<<<dim3(4,16), blk, 0, stream>>>(ws + OFF_ACT2, EXP_WS, 1280, dw3, 655360, db3, 512,
      ws + OFF_DEC, EXP_WS, 512, 0, 1280, 0, 0);

  // elbo per expert, then bin-select
  elbo_kernel<<<dim3(16), dim3(128), 0, stream>>>(v_t, ws);
  select_kernel<<<dim3(1), dim3(128), 0, stream>>>(t, ws, out);
}
